// RHGATBase_28054726377494
// MI455X (gfx1250) — matrix operand order repaired and checked by reading, not compiled, not run on hardware
//
#include <hip/hip_runtime.h>
#include <hip/hip_bf16.h>

typedef _Float16 v8h  __attribute__((ext_vector_type(8)));
typedef _Float16 v16h __attribute__((ext_vector_type(16)));
typedef float    v8f  __attribute__((ext_vector_type(8)));

#define GCND 128   // H*D
#define ENTD 400   // ENT_DIM
#define KPAD 416   // ENT_DIM padded to multiple of 32

// ---------------------------------------------------------------------------
// helpers
// ---------------------------------------------------------------------------
__device__ __forceinline__ void atomicMaxF(float* addr, float v) {
  if (v >= 0.f) atomicMax((int*)addr, __float_as_int(v));
  else          atomicMin((unsigned int*)addr, __float_as_uint(v));
}

__device__ __forceinline__ void atomicAddF(float* addr, float v) {
  __hip_atomic_fetch_add(addr, v, __ATOMIC_RELAXED, __HIP_MEMORY_SCOPE_AGENT);
}

__device__ __forceinline__ v16h cat8(v8h lo, v8h hi) {
  return __builtin_shufflevector(lo, hi, 0,1,2,3,4,5,6,7,8,9,10,11,12,13,14,15);
}

// ---------------------------------------------------------------------------
// 0) prep: W (400x128 f32) -> Wt (128x416 f16, transposed, zero padded)
// ---------------------------------------------------------------------------
__global__ void wt_prep_kernel(const float* __restrict__ W, _Float16* __restrict__ Wt) {
  int idx = blockIdx.x * blockDim.x + threadIdx.x;
  if (idx >= GCND * KPAD) return;
  int n = idx / KPAD;
  int k = idx - n * KPAD;
  Wt[idx] = (k < ENTD) ? (_Float16)W[(size_t)k * GCND + n] : (_Float16)0.f;
}

// ---------------------------------------------------------------------------
// 1) init: zero x accumulator, -inf node max, zero denominators
// ---------------------------------------------------------------------------
__global__ void init_kernel(float* __restrict__ xout, float* __restrict__ nodeMax,
                            float* __restrict__ den, int N_) {
  int idx = blockIdx.x * blockDim.x + threadIdx.x;
  if (idx < N_ * GCND) xout[idx] = 0.f;
  if (idx < N_ * 2) { nodeMax[idx] = -__builtin_inff(); den[idx] = 0.f; }
}

// ---------------------------------------------------------------------------
// 2) tiny GEMMs: r_proj = init_rel @ Wr ; r_out = init_rel @ Wrel
// ---------------------------------------------------------------------------
__global__ void rel_gemm_kernel(const float* __restrict__ init_rel,
                                const float* __restrict__ Wr,
                                const float* __restrict__ Wrel,
                                float* __restrict__ r_proj, float* __restrict__ r_out,
                                int NR2_) {
  int idx = blockIdx.x * blockDim.x + threadIdx.x;
  int total = NR2_ * GCND;
  if (idx >= 2 * total) return;
  int which = idx / total;
  int rc = idx - which * total;
  int row = rc >> 7, col = rc & 127;
  const float* M_ = which ? Wrel : Wr;
  const float* a  = init_rel + (size_t)row * ENTD;
  float s = 0.f;
  #pragma unroll 4
  for (int k = 0; k < ENTD; ++k) s += a[k] * M_[(size_t)k * GCND + col];
  (which ? r_out : r_proj)[rc] = s;
}

// ---------------------------------------------------------------------------
// 3) main GEMM with gather: h[N][128] = init_embed @ W  (f16 WMMA, f32 acc)
//    workgroup: 256 thr = 8 waves; tile 32 rows x 128 cols; wave = 16x32
// ---------------------------------------------------------------------------
__global__ __launch_bounds__(256) void h_gemm_kernel(
    const float* __restrict__ id_embed, const int* __restrict__ ent_feature,
    const float* __restrict__ gender_tab, const float* __restrict__ age_tab,
    const float* __restrict__ level_tab, const _Float16* __restrict__ Wt,
    float* __restrict__ hout, int N_) {
  __shared__ __align__(16) _Float16 ldsA[32][424];   // 424 stride: 16B-aligned rows

  const int tid = threadIdx.x;
  const int rowBase = blockIdx.x * 32;

  // stage gathered A tile (32 rows x 416 cols) as f16
  for (int idx = tid; idx < 32 * KPAD; idx += 256) {
    int r = idx / KPAD;
    int c = idx - r * KPAD;
    int grow = rowBase + r; if (grow >= N_) grow = N_ - 1;
    float v = 0.f;
    if (c < 100)      v = id_embed[(size_t)grow * 100 + c];
    else if (c < 200) v = gender_tab[(size_t)ent_feature[grow * 3 + 0] * 100 + (c - 100)];
    else if (c < 300) v = age_tab   [(size_t)ent_feature[grow * 3 + 1] * 100 + (c - 200)];
    else if (c < 400) v = level_tab [(size_t)ent_feature[grow * 3 + 2] * 100 + (c - 300)];
    ldsA[r][c] = (_Float16)v;
  }
  __syncthreads();

  const int lane = tid & 31;
  const int wave = tid >> 5;
  const int wm = wave >> 2;          // 0..1 : row half of the 32-row tile
  const int wn = wave & 3;           // 0..3 : 32-col slab
  const int halfSel = (lane >= 16) ? 1 : 0;
  const int mrow = lane & 15;

  const int aRow = wm * 16 + mrow;
  const int kOff = halfSel * 8;      // A fragment K offset per ISA layout
  const int kHi  = halfSel * 8;      // B fragment K offset per ISA layout
  const int n0 = wn * 32 + mrow;     // B subtile 0 column
  const int n1 = n0 + 16;            // B subtile 1 column

  v8f acc0 = {}; v8f acc1 = {};

  #pragma unroll
  for (int ks = 0; ks < KPAD / 32; ++ks) {
    const int kbase = ks * 32;
    // A fragment: 16-bit A 16x32 layout (two 16B LDS loads)
    v8h alo = *(const v8h*)&ldsA[aRow][kbase + kOff];
    v8h ahi = *(const v8h*)&ldsA[aRow][kbase + kOff + 16];
    v16h afrag = cat8(alo, ahi);
    // B fragments: 16-bit B 32x16 layout, Wt row-contiguous in K
    const _Float16* b0p = &Wt[(size_t)n0 * KPAD + kbase + kHi];
    const _Float16* b1p = &Wt[(size_t)n1 * KPAD + kbase + kHi];
    v16h b0 = cat8(*(const v8h*)b0p, *(const v8h*)(b0p + 16));
    v16h b1 = cat8(*(const v8h*)b1p, *(const v8h*)(b1p + 16));
    acc0 = __builtin_amdgcn_wmma_f32_16x16x32_f16(false, afrag, false, b0, (short)0, acc0, false, false);
    acc1 = __builtin_amdgcn_wmma_f32_16x16x32_f16(false, afrag, false, b1, (short)0, acc1, false, false);
  }

  // C/D layout: lane L, VGPR v -> row = v + 8*halfSel, col = L%16
  const int outRowBase = rowBase + wm * 16 + halfSel * 8;
  #pragma unroll
  for (int v = 0; v < 8; ++v) {
    int orow = outRowBase + v;
    if (orow < N_) {
      hout[(size_t)orow * GCND + wn * 32 + mrow]      = acc0[v];
      hout[(size_t)orow * GCND + wn * 32 + 16 + mrow] = acc1[v];
    }
  }
}

// ---------------------------------------------------------------------------
// 4) per-edge logits + per-(dst,head) running max. One wave per edge.
// ---------------------------------------------------------------------------
__global__ __launch_bounds__(256) void edge_logits_kernel(
    const int* __restrict__ edge_index, const int* __restrict__ edge_type,
    const float* __restrict__ h, const float* __restrict__ r_proj,
    const float* __restrict__ att_src, const float* __restrict__ att_dst,
    float* __restrict__ logits, float* __restrict__ nodeMax, int E_) {
  const int gw   = (blockIdx.x * blockDim.x + threadIdx.x) >> 5;
  const int lane = threadIdx.x & 31;
  if (gw >= E_) return;
  const int src = edge_index[gw];
  const int dst = edge_index[E_ + gw];
  const int et  = edge_type[gw];
  const int c = lane * 4;                 // channel base; head = c/64 = lane>>4
  const float4 hs = *(const float4*)(h      + (size_t)src * GCND + c);
  const float4 rp = *(const float4*)(r_proj + (size_t)et  * GCND + c);
  const float4 hd = *(const float4*)(h      + (size_t)dst * GCND + c);
  const float4 as = *(const float4*)(att_src + c);
  const float4 ad = *(const float4*)(att_dst + c);
  float p = (hs.x + rp.x) * as.x + (hs.y + rp.y) * as.y +
            (hs.z + rp.z) * as.z + (hs.w + rp.w) * as.w +
            hd.x * ad.x + hd.y * ad.y + hd.z * ad.z + hd.w * ad.w;
  // reduce within each 16-lane head group (don't cross heads: no xor 16)
  p += __shfl_xor(p, 1, 32);
  p += __shfl_xor(p, 2, 32);
  p += __shfl_xor(p, 4, 32);
  p += __shfl_xor(p, 8, 32);
  float lg = (p > 0.f) ? p : 0.2f * p;    // leaky_relu, slope 0.2
  if ((lane & 15) == 0) {
    int head = lane >> 4;
    logits[(size_t)gw * 2 + head] = lg;
    atomicMaxF(&nodeMax[(size_t)dst * 2 + head], lg);
  }
}

// ---------------------------------------------------------------------------
// 5) exp(logit - max) and denominator accumulation (in-place over logits)
// ---------------------------------------------------------------------------
__global__ void edge_exp_kernel(const int* __restrict__ edge_index,
                                float* __restrict__ exlog,
                                const float* __restrict__ nodeMax,
                                float* __restrict__ den, int E_) {
  int idx = blockIdx.x * blockDim.x + threadIdx.x;
  if (idx >= 2 * E_) return;
  int e = idx >> 1, head = idx & 1;
  int dst = edge_index[E_ + e];
  float ex = __expf(exlog[idx] - nodeMax[(size_t)dst * 2 + head]);
  exlog[idx] = ex;
  atomicAddF(&den[(size_t)dst * 2 + head], ex);
}

// ---------------------------------------------------------------------------
// 6) weighted scatter-add of messages into x accumulator. One wave per edge.
// ---------------------------------------------------------------------------
__global__ __launch_bounds__(256) void edge_agg_kernel(
    const int* __restrict__ edge_index, const int* __restrict__ edge_type,
    const float* __restrict__ h, const float* __restrict__ r_proj,
    const float* __restrict__ exbuf, const float* __restrict__ den,
    float* __restrict__ xout, int E_) {
  const int gw   = (blockIdx.x * blockDim.x + threadIdx.x) >> 5;
  const int lane = threadIdx.x & 31;
  if (gw >= E_) return;
  const int src = edge_index[gw];
  const int dst = edge_index[E_ + gw];
  const int et  = edge_type[gw];
  const int c = lane * 4;
  const int head = lane >> 4;
  const float alpha = exbuf[(size_t)gw * 2 + head] /
                      (den[(size_t)dst * 2 + head] + 1e-16f);
  const float4 hs = *(const float4*)(h      + (size_t)src * GCND + c);
  const float4 rp = *(const float4*)(r_proj + (size_t)et  * GCND + c);
  float* base = xout + (size_t)dst * GCND + c;
  atomicAddF(base + 0, alpha * (hs.x + rp.x));
  atomicAddF(base + 1, alpha * (hs.y + rp.y));
  atomicAddF(base + 2, alpha * (hs.z + rp.z));
  atomicAddF(base + 3, alpha * (hs.w + rp.w));
}

// ---------------------------------------------------------------------------
// 7) x = tanh(acc) in place; then 8) sub_emb gather
// ---------------------------------------------------------------------------
__global__ void tanh_kernel(float* __restrict__ xout, int n) {
  int idx = blockIdx.x * blockDim.x + threadIdx.x;
  if (idx < n) xout[idx] = tanhf(xout[idx]);
}

__global__ void gather_sub_kernel(const float* __restrict__ xout,
                                  const int* __restrict__ sub,
                                  float* __restrict__ sub_emb, int B_) {
  int idx = blockIdx.x * blockDim.x + threadIdx.x;
  if (idx >= B_ * GCND) return;
  int b = idx >> 7, c = idx & 127;
  sub_emb[idx] = xout[(size_t)sub[b] * GCND + c];
}

// ---------------------------------------------------------------------------
// launcher
// ---------------------------------------------------------------------------
extern "C" void kernel_launch(void* const* d_in, const int* in_sizes, int n_in,
                              void* d_out, int out_size, void* d_ws, size_t ws_size,
                              hipStream_t stream) {
  const int*   edge_index = (const int*)  d_in[0];
  const int*   edge_type  = (const int*)  d_in[1];
  const int*   ent_feat   = (const int*)  d_in[3];
  const int*   sub        = (const int*)  d_in[4];
  const float* id_embed   = (const float*)d_in[7];
  const float* gender_tab = (const float*)d_in[8];
  const float* age_tab    = (const float*)d_in[9];
  const float* level_tab  = (const float*)d_in[10];
  const float* init_rel   = (const float*)d_in[11];
  const float* W          = (const float*)d_in[12];
  const float* Wr         = (const float*)d_in[13];
  const float* att_src    = (const float*)d_in[14];
  const float* att_dst    = (const float*)d_in[15];
  const float* Wrel       = (const float*)d_in[16];

  const int E_   = in_sizes[1];          // edges
  const int N_   = in_sizes[7] / 100;    // nodes (INIT_DIM = 100)
  const int B_   = in_sizes[4];          // batch
  const int NR2_ = in_sizes[11] / ENTD;  // relation count

  // output layout: sub_emb (B,128) | r (NR2,128) | x (N,128)
  float* out     = (float*)d_out;
  float* sub_emb = out;
  float* r_out   = out + (size_t)B_ * GCND;
  float* x_out   = r_out + (size_t)NR2_ * GCND;

  // workspace layout (floats): h | r_proj | exbuf | nodeMax | den | Wt(f16)
  float* ws      = (float*)d_ws;
  float* h_buf   = ws;
  float* r_proj  = h_buf  + (size_t)N_ * GCND;
  float* exbuf   = r_proj + (size_t)NR2_ * GCND;
  float* nodeMax = exbuf  + (size_t)E_ * 2;
  float* den     = nodeMax + (size_t)N_ * 2;
  _Float16* Wt   = (_Float16*)(den + (size_t)N_ * 2);

  // 0) W -> f16 transposed
  wt_prep_kernel<<<(GCND * KPAD + 255) / 256, 256, 0, stream>>>(W, Wt);
  // 1) init accumulators / stats
  init_kernel<<<((size_t)N_ * GCND + 255) / 256, 256, 0, stream>>>(x_out, nodeMax, den, N_);
  // 2) tiny relation GEMMs
  rel_gemm_kernel<<<(2 * NR2_ * GCND + 255) / 256, 256, 0, stream>>>(
      init_rel, Wr, Wrel, r_proj, r_out, NR2_);
  // 3) h = gathered init_embed @ W  (WMMA)
  h_gemm_kernel<<<(N_ + 31) / 32, 256, 0, stream>>>(
      id_embed, ent_feat, gender_tab, age_tab, level_tab, Wt, h_buf, N_);
  // 4) attention logits + per-node max
  edge_logits_kernel<<<(E_ + 7) / 8, 256, 0, stream>>>(
      edge_index, edge_type, h_buf, r_proj, att_src, att_dst, exbuf, nodeMax, E_);
  // 5) exp + denominators
  edge_exp_kernel<<<(2 * E_ + 255) / 256, 256, 0, stream>>>(
      edge_index, exbuf, nodeMax, den, E_);
  // 6) weighted aggregation into x accumulator
  edge_agg_kernel<<<(E_ + 7) / 8, 256, 0, stream>>>(
      edge_index, edge_type, h_buf, r_proj, exbuf, den, x_out, E_);
  // 7) tanh in place
  tanh_kernel<<<((size_t)N_ * GCND + 255) / 256, 256, 0, stream>>>(x_out, N_ * GCND);
  // 8) gather batch rows
  gather_sub_kernel<<<(B_ * GCND + 255) / 256, 256, 0, stream>>>(x_out, sub, sub_emb, B_);
}
